// BambaAttentionDecoderLayer_34892314313356
// MI455X (gfx1250) — matrix operand order repaired and checked by reading, not compiled, not run on hardware
//
#include <hip/hip_runtime.h>
#include <hip/hip_bf16.h>

// ---------------------------------------------------------------------------
// CDNA5 (gfx1250) decoder layer: RMSNorm -> QKV -> RoPE -> flash GQA attn
// -> O proj (+residual) -> RMSNorm -> gate/up -> SiLU -> down.
// All GEMMs via v_wmma_f32_16x16x32_bf16 (wave32), fp32 accumulate.
// ---------------------------------------------------------------------------

typedef __attribute__((ext_vector_type(16))) __bf16 v16bf;
typedef __attribute__((ext_vector_type(8)))  float  v8f;

#define SEQ   2048
#define HDIM  4096
#define NQH   32
#define NKVH  8
#define HD    128
#define IDIM  14336
#define QKVN  6144   // (32 + 16) * 128

// ---- bf16 <-> f32 (bit ops) ------------------------------------------------
__device__ __forceinline__ unsigned short f2bf(float f) {
  union { float f; unsigned int u; } v; v.f = f;
  unsigned int u = v.u;
  u += 0x7FFFu + ((u >> 16) & 1u);          // round to nearest even
  return (unsigned short)(u >> 16);
}
__device__ __forceinline__ float bf2f(unsigned short h) {
  union { unsigned int u; float f; } v; v.u = ((unsigned int)h) << 16;
  return v.f;
}

// ---- WMMA fragment loads from LDS (bf16, 16x16x32 shapes) ------------------
// A matrix (16 x 32): lane L holds row M = L%16.
//   lane<16 : K = {koff+0..7, koff+16..23};  lane>=16 : K = {koff+8..15, koff+24..31}
__device__ __forceinline__ v16bf lds_frag_a(const unsigned short* s, int rowbase,
                                            int koff, int stride, int lane) {
  const unsigned short* p = s + (rowbase + (lane & 15)) * stride + koff + ((lane & 16) ? 8 : 0);
  union { v16bf v; uint4 q[2]; } u;
  u.q[0] = *(const uint4*)(p);
  u.q[1] = *(const uint4*)(p + 16);
  return u.v;
}
// B matrix (32 x 16): lane L holds col N = L%16; same K order as A:
//   lane<16 : K = {koff+0..7, koff+16..23};  lane>=16 : K = {koff+8..15, koff+24..31}
// (LDS tile stored "N-major, K contiguous")
__device__ __forceinline__ v16bf lds_frag_b(const unsigned short* s, int rowbase,
                                            int koff, int stride, int lane) {
  const unsigned short* p = s + (rowbase + (lane & 15)) * stride + koff + ((lane & 16) ? 8 : 0);
  union { v16bf v; uint4 q[2]; } u;
  u.q[0] = *(const uint4*)(p);
  u.q[1] = *(const uint4*)(p + 16);
  return u.v;
}
__device__ __forceinline__ v8f wmma_bf16(v16bf a, v16bf b, v8f c) {
  return __builtin_amdgcn_wmma_f32_16x16x32_bf16(false, a, false, b, (short)0, c,
                                                 false, false);
}

// ---------------------------------------------------------------------------
// RMSNorm: one block per row (4096 cols), writes bf16.
// ---------------------------------------------------------------------------
__global__ __launch_bounds__(256) void rmsnorm_kernel(const float* __restrict__ x,
                                                      const float* __restrict__ w,
                                                      unsigned short* __restrict__ out) {
  const int row = blockIdx.x;
  const float* xr = x + (size_t)row * HDIM;
  float ss = 0.f;
  for (int i = threadIdx.x; i < HDIM / 4; i += 256) {
    float4 v = ((const float4*)xr)[i];
    ss += v.x * v.x + v.y * v.y + v.z * v.z + v.w * v.w;
  }
  for (int off = 16; off > 0; off >>= 1) ss += __shfl_xor(ss, off, 32);
  __shared__ float red[8];
  if ((threadIdx.x & 31) == 0) red[threadIdx.x >> 5] = ss;
  __syncthreads();
  float tot = 0.f;
  for (int i = 0; i < 8; ++i) tot += red[i];
  const float inv = rsqrtf(tot / (float)HDIM + 1e-5f);
  unsigned short* orow = out + (size_t)row * HDIM;
  for (int i = threadIdx.x; i < HDIM / 4; i += 256) {
    float4 v = ((const float4*)xr)[i];
    const float4 wv = ((const float4*)w)[i];
    ushort4 o;
    o.x = f2bf(v.x * inv * wv.x);
    o.y = f2bf(v.y * inv * wv.y);
    o.z = f2bf(v.z * inv * wv.z);
    o.w = f2bf(v.w * inv * wv.w);
    ((ushort4*)orow)[i] = o;
  }
}

// ---------------------------------------------------------------------------
// Generic WMMA GEMM: C[M,N] = A_bf16[M,K] * B_f32[K,N]   (B converted to bf16)
// MODE 0: write f32   MODE 1: write f32 + Res   MODE 2: write bf16
// Tile 128x128, BK=32, 256 threads = 8 waves (2x4), each wave does 64x32.
// Double-buffered LDS with register staging; grid = (M/128, N/128) so the 16
// row-tile blocks sharing one weight strip are consecutive -> B stays in L2.
// ---------------------------------------------------------------------------
template <int MODE>
__global__ __launch_bounds__(256) void gemm_bf16_kernel(
    const unsigned short* __restrict__ A, const float* __restrict__ B,
    const float* __restrict__ Res, float* __restrict__ Cf,
    unsigned short* __restrict__ Cb, int M, int N, int K) {
  __shared__ __align__(16) unsigned short sA[2][128 * 48];
  __shared__ __align__(16) unsigned short sB[2][128 * 48];  // N-major, K contig
  const int tid = threadIdx.x;
  const int lane = tid & 31, wid = tid >> 5;
  const int wm = wid & 1, wn = wid >> 1;                    // 2 x 4 wave grid
  const int m0 = blockIdx.x * 128, n0 = blockIdx.y * 128;

  // per-thread staging addresses
  const int ar_r = tid >> 2, ar_c = (tid & 3) * 8;          // A chunk 0 (+64 rows)
  const int br_k = tid >> 5, br_n = (tid & 31) * 4;         // B chunk 0 (+8 k)
  const unsigned short* Ag = A + (size_t)(m0 + ar_r) * K + ar_c;
  const float* Bg0 = B + (size_t)br_k * N + n0 + br_n;

  v8f acc[4][2];
  #pragma unroll
  for (int i = 0; i < 4; ++i)
    #pragma unroll
    for (int j = 0; j < 2; ++j) acc[i][j] = 0.f;

  uint4  ar[2];
  float4 br[4];
  const int nt = K >> 5;

  // ---- prologue: stage tile 0 ----
  ar[0] = *(const uint4*)(Ag);
  ar[1] = *(const uint4*)(Ag + (size_t)64 * K);
  #pragma unroll
  for (int j = 0; j < 4; ++j) br[j] = *(const float4*)(Bg0 + (size_t)(8 * j) * N);
  {
    *(uint4*)&sA[0][ar_r * 48 + ar_c]        = ar[0];
    *(uint4*)&sA[0][(ar_r + 64) * 48 + ar_c] = ar[1];
    #pragma unroll
    for (int j = 0; j < 4; ++j) {
      int kr = br_k + 8 * j;
      sB[0][(br_n + 0) * 48 + kr] = f2bf(br[j].x);
      sB[0][(br_n + 1) * 48 + kr] = f2bf(br[j].y);
      sB[0][(br_n + 2) * 48 + kr] = f2bf(br[j].z);
      sB[0][(br_n + 3) * 48 + kr] = f2bf(br[j].w);
    }
  }

  for (int kt = 0; kt < nt; ++kt) {
    __syncthreads();
    const int cur = kt & 1;
    const bool more = (kt + 1) < nt;
    if (more) {
      const int k0 = (kt + 1) * 32;
      ar[0] = *(const uint4*)(Ag + k0);
      ar[1] = *(const uint4*)(Ag + (size_t)64 * K + k0);
      #pragma unroll
      for (int j = 0; j < 4; ++j)
        br[j] = *(const float4*)(Bg0 + (size_t)(k0 + 8 * j) * N);
      if (kt + 8 < nt)  // prime L2 for the weight strip a few tiles ahead
        __builtin_prefetch(Bg0 + (size_t)((kt + 8) * 32) * N, 0, 1);
    }
    // compute from current buffer
    v16bf bf0 = lds_frag_b(sB[cur], wn * 32 + 0, 0, 48, lane);
    v16bf bf1 = lds_frag_b(sB[cur], wn * 32 + 16, 0, 48, lane);
    #pragma unroll
    for (int mi = 0; mi < 4; ++mi) {
      v16bf af = lds_frag_a(sA[cur], wm * 64 + mi * 16, 0, 48, lane);
      acc[mi][0] = wmma_bf16(af, bf0, acc[mi][0]);
      acc[mi][1] = wmma_bf16(af, bf1, acc[mi][1]);
    }
    if (more) {
      const int nxt = cur ^ 1;
      *(uint4*)&sA[nxt][ar_r * 48 + ar_c]        = ar[0];
      *(uint4*)&sA[nxt][(ar_r + 64) * 48 + ar_c] = ar[1];
      #pragma unroll
      for (int j = 0; j < 4; ++j) {
        int kr = br_k + 8 * j;
        sB[nxt][(br_n + 0) * 48 + kr] = f2bf(br[j].x);
        sB[nxt][(br_n + 1) * 48 + kr] = f2bf(br[j].y);
        sB[nxt][(br_n + 2) * 48 + kr] = f2bf(br[j].z);
        sB[nxt][(br_n + 3) * 48 + kr] = f2bf(br[j].w);
      }
    }
  }

  // Epilogue (branchless per MODE). C frag: M = r + 8*(lane>=16), N = lane%16
  #pragma unroll
  for (int mi = 0; mi < 4; ++mi)
    #pragma unroll
    for (int ni = 0; ni < 2; ++ni)
      #pragma unroll
      for (int r = 0; r < 8; ++r) {
        int row = m0 + wm * 64 + mi * 16 + r + ((lane >> 4) << 3);
        int col = n0 + wn * 32 + ni * 16 + (lane & 15);
        size_t idx = (size_t)row * N + col;
        float v = acc[mi][ni][r];
        if (MODE == 1) v += Res[idx];
        if (MODE == 2) Cb[idx] = f2bf(v);
        else           Cf[idx] = v;
      }
}

// ---------------------------------------------------------------------------
// RoPE + pack: qkv f32 [S,6144] -> Q bf16 [32][S][128], K bf16 [8][S][128],
// V^T bf16 [8][128][S]. One block per token.
// ---------------------------------------------------------------------------
__global__ __launch_bounds__(256) void rope_pack_kernel(
    const int* __restrict__ pos, const float* __restrict__ qkv,
    unsigned short* __restrict__ qb, unsigned short* __restrict__ kb,
    unsigned short* __restrict__ vtb) {
  const int s = blockIdx.x;
  const float p = (float)pos[s];
  const float* row = qkv + (size_t)s * QKVN;
  for (int i = threadIdx.x; i < (NQH + NKVH) * 64; i += 256) {
    int head = i >> 6, d = i & 63;
    float invf = __powf(10000.0f, -(float)d * (1.0f / 64.0f));
    float ang = p * invf;
    float c = __cosf(ang), sn = __sinf(ang);
    const float* base;
    unsigned short* ob;
    if (head < NQH) {
      base = row + head * HD;
      ob = qb + ((size_t)head * SEQ + s) * HD;
    } else {
      int kh = head - NQH;
      base = row + NQH * HD + kh * HD;
      ob = kb + ((size_t)kh * SEQ + s) * HD;
    }
    float x1 = base[d], x2 = base[d + 64];
    ob[d]      = f2bf(x1 * c - x2 * sn);
    ob[d + 64] = f2bf(x2 * c + x1 * sn);
  }
  for (int i = threadIdx.x; i < NKVH * HD; i += 256) {
    int kh = i >> 7, d = i & 127;
    vtb[((size_t)kh * HD + d) * SEQ + s] =
        f2bf(row[(NQH + NKVH) * HD + kh * HD + d]);
  }
}

// ---------------------------------------------------------------------------
// Flash attention, causal GQA. Block = (q-tile of 128, one q head).
// 8 waves; wave w owns q rows [w*16, w*16+16). K tiles of 32 keys.
// ---------------------------------------------------------------------------
__global__ __launch_bounds__(256) void flash_attn_kernel(
    const unsigned short* __restrict__ qb, const unsigned short* __restrict__ kb,
    const unsigned short* __restrict__ vtb, unsigned short* __restrict__ attn) {
  constexpr int QS = HD + 16;  // 144 elem stride (288B, 16B-aligned frag reads)
  constexpr int PS = 48;       // 32 + 16 pad
  __shared__ __align__(16) unsigned short sQ[128 * QS];
  __shared__ __align__(16) unsigned short sK[32 * QS];
  __shared__ __align__(16) unsigned short sV[128 * PS];   // V^T tile: [d][k]
  __shared__ __align__(16) unsigned short sP[128 * PS];

  const int h = blockIdx.y, kh = h >> 2;
  const int q0 = blockIdx.x * 128;
  const int tid = threadIdx.x, lane = tid & 31, w = tid >> 5;

  const unsigned short* Qg = qb + ((size_t)h * SEQ + q0) * HD;
  #pragma unroll
  for (int it = 0; it < 8; ++it) {
    int i = tid + it * 256;
    int r = i >> 4, c = (i & 15) * 8;
    *(uint4*)&sQ[r * QS + c] = *(const uint4*)(Qg + (size_t)r * HD + c);
  }

  v8f o[8];
  #pragma unroll
  for (int i = 0; i < 8; ++i) o[i] = 0.f;
  float m[8], l[8];
  #pragma unroll
  for (int r = 0; r < 8; ++r) { m[r] = -3.0e38f; l[r] = 0.f; }

  const int nkt = blockIdx.x * 4 + 4;
  const float sc = 0.08838834764831845f;     // 1/sqrt(128)
  const int qrow = q0 + w * 16 + ((lane >> 4) << 3);

  for (int kt = 0; kt < nkt; ++kt) {
    const int k0 = kt * 32;
    __syncthreads();
    #pragma unroll
    for (int it = 0; it < 2; ++it) {
      int i = tid + it * 256;
      int r = i >> 4, c = (i & 15) * 8;
      *(uint4*)&sK[r * QS + c] =
          *(const uint4*)(kb + ((size_t)kh * SEQ + k0 + r) * HD + c);
    }
    #pragma unroll
    for (int it = 0; it < 2; ++it) {
      int i = tid + it * 256;
      int r = i >> 2, c = (i & 3) * 8;
      *(uint4*)&sV[r * PS + c] =
          *(const uint4*)(vtb + ((size_t)kh * HD + r) * SEQ + k0 + c);
    }
    __syncthreads();

    v8f s0 = 0.f, s1 = 0.f;
    #pragma unroll
    for (int kc = 0; kc < 4; ++kc) {
      v16bf aq = lds_frag_a(sQ, w * 16, kc * 32, QS, lane);
      v16bf b0 = lds_frag_b(sK, 0,  kc * 32, QS, lane);
      v16bf b1 = lds_frag_b(sK, 16, kc * 32, QS, lane);
      s0 = wmma_bf16(aq, b0, s0);
      s1 = wmma_bf16(aq, b1, s1);
    }

    const int kc0 = k0 + (lane & 15), kc1 = kc0 + 16;
    #pragma unroll
    for (int r = 0; r < 8; ++r) {
      float a0 = s0[r] * sc; if (kc0 > qrow + r) a0 = -3.0e38f;
      float a1 = s1[r] * sc; if (kc1 > qrow + r) a1 = -3.0e38f;
      float mx = fmaxf(a0, a1);
      mx = fmaxf(mx, __shfl_xor(mx, 1, 16));
      mx = fmaxf(mx, __shfl_xor(mx, 2, 16));
      mx = fmaxf(mx, __shfl_xor(mx, 4, 16));
      mx = fmaxf(mx, __shfl_xor(mx, 8, 16));
      const float mn = fmaxf(m[r], mx);
      const float p0 = __expf(a0 - mn), p1 = __expf(a1 - mn);
      float rs = p0 + p1;
      rs += __shfl_xor(rs, 1, 16);
      rs += __shfl_xor(rs, 2, 16);
      rs += __shfl_xor(rs, 4, 16);
      rs += __shfl_xor(rs, 8, 16);
      const float scale = __expf(m[r] - mn);
      l[r] = l[r] * scale + rs;
      m[r] = mn;
      #pragma unroll
      for (int d8 = 0; d8 < 8; ++d8) o[d8][r] *= scale;
      const int prow = w * 16 + r + ((lane >> 4) << 3);
      sP[prow * PS + (lane & 15)]      = f2bf(p0);
      sP[prow * PS + 16 + (lane & 15)] = f2bf(p1);
    }

    v16bf ap = lds_frag_a(sP, w * 16, 0, PS, lane);
    #pragma unroll
    for (int d8 = 0; d8 < 8; ++d8) {
      v16bf bv = lds_frag_b(sV, d8 * 16, 0, PS, lane);
      o[d8] = wmma_bf16(ap, bv, o[d8]);
    }
  }

  #pragma unroll
  for (int r = 0; r < 8; ++r) {
    const float invl = 1.0f / l[r];
    const int q = qrow + r;
    #pragma unroll
    for (int d8 = 0; d8 < 8; ++d8)
      attn[(size_t)q * (NQH * HD) + h * HD + d8 * 16 + (lane & 15)] =
          f2bf(o[d8][r] * invl);
  }
}

// ---------------------------------------------------------------------------
// h = silu(gate) * up   (gu bf16 [S, 2*I] -> h bf16 [S, I])
// ---------------------------------------------------------------------------
__global__ __launch_bounds__(256) void silu_mul_kernel(
    const unsigned short* __restrict__ gu, unsigned short* __restrict__ h) {
  const int col = blockIdx.x * 256 + threadIdx.x;   // < 14336
  const int row = blockIdx.y;
  const float g = bf2f(gu[(size_t)row * (2 * IDIM) + col]);
  const float u = bf2f(gu[(size_t)row * (2 * IDIM) + IDIM + col]);
  h[(size_t)row * IDIM + col] = f2bf(g * (1.0f / (1.0f + __expf(-g))) * u);
}

// ---------------------------------------------------------------------------
extern "C" void kernel_launch(void* const* d_in, const int* in_sizes, int n_in,
                              void* d_out, int out_size, void* d_ws, size_t ws_size,
                              hipStream_t stream) {
  (void)in_sizes; (void)n_in; (void)out_size; (void)ws_size;
  const int*   positions = (const int*)d_in[0];
  const float* hidden    = (const float*)d_in[1];
  const float* w_qkv     = (const float*)d_in[2];
  const float* w_o       = (const float*)d_in[3];
  const float* w_gate_up = (const float*)d_in[4];
  const float* w_down    = (const float*)d_in[5];
  const float* ln1_w     = (const float*)d_in[6];
  const float* ln2_w     = (const float*)d_in[7];

  float* out_mlp  = (float*)d_out;                                // output 0
  float* residual = (float*)d_out + (size_t)SEQ * HDIM;           // output 1

  char* ws = (char*)d_ws;
  unsigned short* xb   = (unsigned short*)(ws + 0);                        // 16 MB
  float*          qkvf = (float*)(ws + (size_t)16 * 1024 * 1024);          // 48 MB
  unsigned short* qb   = (unsigned short*)(ws + (size_t)64 * 1024 * 1024); // 16 MB
  unsigned short* kb   = (unsigned short*)(ws + (size_t)80 * 1024 * 1024); //  4 MB
  unsigned short* vtb  = (unsigned short*)(ws + (size_t)84 * 1024 * 1024); //  4 MB
  unsigned short* attn = (unsigned short*)(ws + (size_t)88 * 1024 * 1024); // 16 MB
  unsigned short* gu   = (unsigned short*)(ws + (size_t)104 * 1024 * 1024);//112 MB
  unsigned short* hb   = (unsigned short*)(ws + (size_t)216 * 1024 * 1024);// 56 MB

  // 1) x = rmsnorm(hidden, ln1) -> bf16
  rmsnorm_kernel<<<SEQ, 256, 0, stream>>>(hidden, ln1_w, xb);
  // 2) qkv = x @ w_qkv  (f32 out); grid (M-tiles, N-tiles) for L2 weight reuse
  gemm_bf16_kernel<0><<<dim3(SEQ / 128, QKVN / 128), 256, 0, stream>>>(
      xb, w_qkv, nullptr, qkvf, nullptr, SEQ, QKVN, HDIM);
  // 3) rope + pack q/k/v^T
  rope_pack_kernel<<<SEQ, 256, 0, stream>>>(positions, qkvf, qb, kb, vtb);
  // 4) flash attention -> attn bf16 [S, 4096]
  flash_attn_kernel<<<dim3(SEQ / 128, NQH), 256, 0, stream>>>(qb, kb, vtb, attn);
  // 5) residual = attn @ w_o + hidden  (f32, second output)
  gemm_bf16_kernel<1><<<dim3(SEQ / 128, HDIM / 128), 256, 0, stream>>>(
      attn, w_o, hidden, residual, nullptr, SEQ, HDIM, NQH * HD);
  // 6) x2 = rmsnorm(residual, ln2) -> bf16 (reuse xb)
  rmsnorm_kernel<<<SEQ, 256, 0, stream>>>(residual, ln2_w, xb);
  // 7) gu = x2 @ w_gate_up (bf16 out)
  gemm_bf16_kernel<2><<<dim3(SEQ / 128, (2 * IDIM) / 128), 256, 0, stream>>>(
      xb, w_gate_up, nullptr, nullptr, gu, SEQ, 2 * IDIM, HDIM);
  // 8) h = silu(gate) * up
  silu_mul_kernel<<<dim3(IDIM / 256, SEQ), 256, 0, stream>>>(gu, hb);
  // 9) out = h @ w_down (f32, first output)
  gemm_bf16_kernel<0><<<dim3(SEQ / 128, HDIM / 128), 256, 0, stream>>>(
      hb, w_down, nullptr, out_mlp, nullptr, SEQ, HDIM, IDIM);
}
